// Informer_82386062672134
// MI455X (gfx1250) — compile-verified, matrix operand order checked by reading, not run on hardware
//
#include <hip/hip_runtime.h>
#include <hip/hip_bf16.h>

// ---------------------------------------------------------------------------
// Informer forward for MI455X (gfx1250): all heavy GEMMs + attention through
// v_wmma_f32_16x16x32_f16 (f32 -> f16 staged in LDS, f32 accumulate).
// Staging uses float4 global loads + packed-half LDS stores; K-chunk = 64.
// ---------------------------------------------------------------------------

typedef __attribute__((ext_vector_type(16))) _Float16 v16h;
typedef __attribute__((ext_vector_type(8)))  float    v8f;

#define DM    512
#define NH    8
#define HD    64
#define DFF   2048
#define BATCH 4

__device__ __forceinline__ unsigned pack2h(float a, float b) {
  union { _Float16 h[2]; unsigned u; } u;
  u.h[0] = (_Float16)a; u.h[1] = (_Float16)b;
  return u.u;
}

// Load one 16-lane-row x 32-k f16 fragment (A or B operand) from LDS.
// Packing per CDNA5 ISA 7.12.2 (16-bit A 16x32): lanes 0-15 rows, hi=lane>>4
// selects the K sub-block; pairs contiguous -> compiler emits ds_load_b128 x2.
__device__ __forceinline__ v16h frag_k32(const _Float16* p, int hi) {
  v16h f;
#pragma unroll
  for (int v = 0; v < 8; ++v) {
    int kk = (v < 4) ? (hi * 8 + v * 2) : (16 + hi * 8 + (v - 4) * 2);
    f[2 * v]     = p[kk];
    f[2 * v + 1] = p[kk + 1];
  }
  return f;
}

__device__ __forceinline__ float gelu_exact(float x) {
  return 0.5f * x * (1.0f + erff(x * 0.70710678118f));
}

// ---------------------------------------------------------------------------
// GEMM: C[M,N] = act(A[M,K] @ W[K,N] + bias (+C if accum))
// A row r lives at A + (r / aRPB)*aBS + (r % aRPB)*K   (circular-pad support)
// W element (k,n) lives at W[k*wRS + n*wCS + wOff]     (conv-weight support)
// Block tile 128x64, 256 threads (8 waves), wave tile 32x32 (2x2 WMMA),
// staged K-chunk 64 (2 WMMA k-steps per barrier).
// ---------------------------------------------------------------------------
__global__ __launch_bounds__(256)
void gemm_wmma(const float* __restrict__ A, const float* __restrict__ W,
               const float* __restrict__ bias, float* __restrict__ C,
               int M, int N, int K, int act, int accum,
               int aRPB, long aBS, int wRS, int wCS, long wOff) {
  __shared__ _Float16 sA[128 * 64];   // [m][k]
  __shared__ _Float16 sB[64 * 64];    // [n][k] (transposed so k-pairs contiguous)

  const int t    = threadIdx.x;
  const int lane = t & 31, w = t >> 5;
  const int wm   = w & 3,  wn = w >> 2;
  const int m0   = blockIdx.y * 128;
  const int n0   = blockIdx.x * 64;
  const int hi   = lane >> 4, lr = lane & 15;

  v8f acc[2][2];
#pragma unroll
  for (int mi = 0; mi < 2; ++mi)
#pragma unroll
    for (int ni = 0; ni < 2; ++ni)
#pragma unroll
      for (int i = 0; i < 8; ++i) acc[mi][ni][i] = 0.0f;

  for (int k0 = 0; k0 < K; k0 += 64) {
    // ---- Stage A tile 128x64: 8 float4 loads/thread, packed b64 LDS stores
#pragma unroll
    for (int i = 0; i < 8; ++i) {
      int e  = t + i * 256;          // 0..2047 float4 slots
      int r  = e >> 4;               // row 0..127
      int c4 = (e & 15) * 4;         // k offset 0..60
      int gr = m0 + r;
      const float* ap = (aBS == 0)
          ? A + (long)gr * K + (k0 + c4)
          : A + (long)(gr / aRPB) * aBS + (long)(gr % aRPB) * K + (k0 + c4);
      float4 f = *(const float4*)ap;
      uint2 p;
      p.x = pack2h(f.x, f.y);
      p.y = pack2h(f.z, f.w);
      *(uint2*)&sA[r * 64 + c4] = p;
    }
    // ---- Stage B tile 64x64 -> [n][k]: units of (2k x 4n)
#pragma unroll
    for (int i = 0; i < 2; ++i) {
      int e = t + i * 256;           // 0..511 units
      int k = (e >> 4) * 2;          // even k
      int n = (e & 15) * 4;
      if (wCS == 1) {                // dense row-major weights: float4 loads
        const float* w0 = W + (long)(k0 + k) * wRS + (n0 + n);
        const float* w1 = w0 + wRS;
        float4 fa = *(const float4*)w0;
        float4 fb = *(const float4*)w1;
        *(unsigned*)&sB[(n + 0) * 64 + k] = pack2h(fa.x, fb.x);
        *(unsigned*)&sB[(n + 1) * 64 + k] = pack2h(fa.y, fb.y);
        *(unsigned*)&sB[(n + 2) * 64 + k] = pack2h(fa.z, fb.z);
        *(unsigned*)&sB[(n + 3) * 64 + k] = pack2h(fa.w, fb.w);
      } else {                       // strided conv weights
        for (int j = 0; j < 4; ++j) {
          float x0 = W[(long)(k0 + k) * wRS + (long)(n0 + n + j) * wCS + wOff];
          float x1 = W[(long)(k0 + k + 1) * wRS + (long)(n0 + n + j) * wCS + wOff];
          *(unsigned*)&sB[(n + j) * 64 + k] = pack2h(x0, x1);
        }
      }
    }
    // ---- Prefetch next K-chunk into L2 (global_prefetch_b8)
    if (k0 + 64 < K) {
      int gr = m0 + (t >> 1);
      const float* ap = (aBS == 0)
          ? A + (long)gr * K + (k0 + 64)
          : A + (long)(gr / aRPB) * aBS + (long)(gr % aRPB) * K + (k0 + 64);
      __builtin_prefetch(ap, 0, 1);
      __builtin_prefetch(
          &W[(long)(k0 + 64 + (t >> 2)) * wRS + (long)(n0 + (t & 3) * 16) * wCS + wOff],
          0, 1);
    }
    __syncthreads();

#pragma unroll
    for (int ks = 0; ks < 2; ++ks) {
      v16h a0 = frag_k32(&sA[(wm * 32 + lr) * 64 + ks * 32], hi);
      v16h a1 = frag_k32(&sA[(wm * 32 + 16 + lr) * 64 + ks * 32], hi);
      v16h b0 = frag_k32(&sB[(wn * 32 + lr) * 64 + ks * 32], hi);
      v16h b1 = frag_k32(&sB[(wn * 32 + 16 + lr) * 64 + ks * 32], hi);
      acc[0][0] = __builtin_amdgcn_wmma_f32_16x16x32_f16(false, a0, false, b0, (short)0, acc[0][0], false, false);
      acc[0][1] = __builtin_amdgcn_wmma_f32_16x16x32_f16(false, a0, false, b1, (short)0, acc[0][1], false, false);
      acc[1][0] = __builtin_amdgcn_wmma_f32_16x16x32_f16(false, a1, false, b0, (short)0, acc[1][0], false, false);
      acc[1][1] = __builtin_amdgcn_wmma_f32_16x16x32_f16(false, a1, false, b1, (short)0, acc[1][1], false, false);
    }
    __syncthreads();
  }

  // Epilogue: C/D layout -> row = i + 8*hi (per 16-tile), col = lr
#pragma unroll
  for (int mi = 0; mi < 2; ++mi)
#pragma unroll
    for (int ni = 0; ni < 2; ++ni)
#pragma unroll
      for (int i = 0; i < 8; ++i) {
        int gr = m0 + wm * 32 + mi * 16 + i + 8 * hi;
        int gc = n0 + wn * 32 + ni * 16 + lr;
        float v = acc[mi][ni][i];
        if (bias)  v += bias[gc];
        if (accum) v += C[(long)gr * N + gc];
        if (act == 1) v = gelu_exact(v);
        C[(long)gr * N + gc] = v;
      }
}

// ---------------------------------------------------------------------------
// Fused attention (flash-style): one block = 64 queries x one head.
// Q,K,V: [B, L, 512] (head-interleaved). O: [B, H, Lq, 64].
// 128 threads = 4 waves; wave w owns query rows w*16..w*16+15.
// ---------------------------------------------------------------------------
__global__ __launch_bounds__(128)
void attn_wmma(const float* __restrict__ Q, const float* __restrict__ Kx,
               const float* __restrict__ Vx, float* __restrict__ O,
               int Lq, int Lk, int causal) {
  __shared__ _Float16 sQ[64 * 64];     // [m][d]
  __shared__ _Float16 sK[64 * 64];     // [n][d]
  __shared__ _Float16 sVt[64 * 64];    // [d][key]
  __shared__ _Float16 sP[4][16 * 64];  // per-wave P tile [m][key]
  __shared__ float    sS[4][16 * 64];  // per-wave S tile
  __shared__ float    sM[64], sL[64], sAl[64];

  const int b = blockIdx.z, h = blockIdx.y, qt = blockIdx.x;
  const int t = threadIdx.x, lane = t & 31, w = t >> 5;
  const int hi = lane >> 4, lr = lane & 15;

#pragma unroll
  for (int i = 0; i < 8; ++i) {
    int e = t + i * 128;               // 0..1023 float4 slots
    int m = e >> 4, d4 = (e & 15) * 4;
    float4 f = *(const float4*)&Q[((long)b * Lq + qt * 64 + m) * DM + h * HD + d4];
    uint2 p; p.x = pack2h(f.x, f.y); p.y = pack2h(f.z, f.w);
    *(uint2*)&sQ[m * 64 + d4] = p;
  }
  if (t < 64) { sM[t] = -1e30f; sL[t] = 0.0f; }

  v8f oacc[4];
#pragma unroll
  for (int ni = 0; ni < 4; ++ni)
#pragma unroll
    for (int i = 0; i < 8; ++i) oacc[ni][i] = 0.0f;
  __syncthreads();

  const int ktEnd = causal ? qt : (Lk / 64 - 1);
  for (int kt = 0; kt <= ktEnd; ++kt) {
#pragma unroll
    for (int i = 0; i < 8; ++i) {
      int e = t + i * 128;
      int n = e >> 4, d4 = (e & 15) * 4;
      long src = ((long)b * Lk + kt * 64 + n) * DM + h * HD + d4;
      float4 kf = *(const float4*)&Kx[src];
      float4 vf = *(const float4*)&Vx[src];
      uint2 p; p.x = pack2h(kf.x, kf.y); p.y = pack2h(kf.z, kf.w);
      *(uint2*)&sK[n * 64 + d4] = p;
      sVt[(d4 + 0) * 64 + n] = (_Float16)vf.x;
      sVt[(d4 + 1) * 64 + n] = (_Float16)vf.y;
      sVt[(d4 + 2) * 64 + n] = (_Float16)vf.z;
      sVt[(d4 + 3) * 64 + n] = (_Float16)vf.w;
    }
    if (kt < ktEnd) {                  // prefetch next key tile
      long nsrc = ((long)b * Lk + (kt + 1) * 64 + (t >> 1)) * DM + h * HD + (t & 1) * 32;
      __builtin_prefetch(&Kx[nsrc], 0, 1);
      __builtin_prefetch(&Vx[nsrc], 0, 1);
    }
    __syncthreads();

    // S = Q K^T (k-dim = head dim 64 -> 2 WMMA k-steps)
    v8f sacc[4];
#pragma unroll
    for (int ni = 0; ni < 4; ++ni)
#pragma unroll
      for (int i = 0; i < 8; ++i) sacc[ni][i] = 0.0f;
#pragma unroll
    for (int ks = 0; ks < 2; ++ks) {
      v16h a = frag_k32(&sQ[(w * 16 + lr) * 64 + ks * 32], hi);
#pragma unroll
      for (int ni = 0; ni < 4; ++ni) {
        v16h bf = frag_k32(&sK[(ni * 16 + lr) * 64 + ks * 32], hi);
        sacc[ni] = __builtin_amdgcn_wmma_f32_16x16x32_f16(false, a, false, bf, (short)0, sacc[ni], false, false);
      }
    }
#pragma unroll
    for (int ni = 0; ni < 4; ++ni)
#pragma unroll
      for (int i = 0; i < 8; ++i)
        sS[w][(i + 8 * hi) * 64 + ni * 16 + lr] = sacc[ni][i] * 0.125f;
    __syncthreads();

    // Online softmax: lanes (lr, lr+16) split row lr's 64 cols.
    const int r  = lr;
    const int gq = qt * 64 + w * 16 + r;
    float tmax = -1e30f;
    for (int j = 0; j < 32; ++j) {
      int cc = hi * 32 + j;
      float s = sS[w][r * 64 + cc];
      if (causal && (kt * 64 + cc) > gq) s = -1e30f;
      tmax = fmaxf(tmax, s);
    }
    tmax = fmaxf(tmax, __shfl_xor(tmax, 16, 32));
    float mold  = sM[w * 16 + r];
    float mnew  = fmaxf(mold, tmax);
    float alpha = __expf(mold - mnew);
    float psum  = 0.0f;
    for (int j = 0; j < 32; ++j) {
      int cc = hi * 32 + j;
      float s = sS[w][r * 64 + cc];
      if (causal && (kt * 64 + cc) > gq) s = -1e30f;
      float p = __expf(s - mnew);
      psum += p;
      sP[w][r * 64 + cc] = (_Float16)p;
    }
    psum += __shfl_xor(psum, 16, 32);
    if (hi == 0) {
      sM[w * 16 + r]  = mnew;
      sL[w * 16 + r]  = sL[w * 16 + r] * alpha + psum;
      sAl[w * 16 + r] = alpha;
    }
    __syncthreads();

    // O = alpha*O + P @ V
#pragma unroll
    for (int ni = 0; ni < 4; ++ni)
#pragma unroll
      for (int i = 0; i < 8; ++i) oacc[ni][i] *= sAl[w * 16 + i + 8 * hi];

#pragma unroll
    for (int ks = 0; ks < 2; ++ks) {
      v16h a = frag_k32(&sP[w][lr * 64 + ks * 32], hi);
#pragma unroll
      for (int ni = 0; ni < 4; ++ni) {
        v16h bf = frag_k32(&sVt[(ni * 16 + lr) * 64 + ks * 32], hi);
        oacc[ni] = __builtin_amdgcn_wmma_f32_16x16x32_f16(false, a, false, bf, (short)0, oacc[ni], false, false);
      }
    }
    __syncthreads();
  }

#pragma unroll
  for (int ni = 0; ni < 4; ++ni)
#pragma unroll
    for (int i = 0; i < 8; ++i) {
      int rr = w * 16 + i + 8 * hi;
      float l = sL[rr];
      O[(((long)b * NH + h) * Lq + qt * 64 + rr) * HD + ni * 16 + lr] =
          oacc[ni][i] / l;
    }
}

// ---------------------------------------------------------------------------
// Elementwise / reduction helper kernels
// ---------------------------------------------------------------------------

// Embedding: circular conv1d(Cin->512, k=3) + sinusoidal positional + temporal.
__global__ void embed_kernel(const float* __restrict__ X, const int* __restrict__ Mark,
                             const float* __restrict__ Wc, float* __restrict__ Out,
                             int L, int Cin) {
  long idx = (long)blockIdx.x * 256 + threadIdx.x;
  if (idx >= (long)BATCH * L * DM) return;
  int b = idx / ((long)L * DM);
  int rem = idx % ((long)L * DM);
  int tp = rem / DM, c = rem % DM;
  float acc = 0.0f;
  for (int dt = 0; dt < 3; ++dt) {
    int st = tp - 1 + dt;
    st = st < 0 ? L - 1 : (st >= L ? 0 : st);
    const float* xr = X + ((long)b * L + st) * Cin;
    for (int i = 0; i < Cin; ++i) acc += xr[i] * Wc[c * Cin * 3 + i * 3 + dt];
  }
  float div = __expf(-(float)(c & ~1) * 0.017988946f);  // ln(10000)/512
  float posv = (c & 1) ? __cosf(tp * div) : __sinf(tp * div);
  const int* mk = Mark + ((long)b * L + tp) * 4;
  float tempv = 0.0f;
  for (int j = 0; j < 4; ++j) {
    float p = (float)mk[j];
    tempv += (c & 1) ? __cosf(p * div) : __sinf(p * div);
  }
  Out[idx] = acc + posv + tempv;
}

// LayerNorm(X + R) * G + B  over last dim (512). One block per row.
__global__ __launch_bounds__(128)
void ln_kernel(const float* __restrict__ X, const float* __restrict__ R,
               const float* __restrict__ G, const float* __restrict__ Bt,
               float* __restrict__ Y) {
  __shared__ float red[128];
  int row = blockIdx.x, t = threadIdx.x;
  const float* xr = X + (long)row * DM;
  const float* rr = R ? R + (long)row * DM : nullptr;
  float v[4];
  float s = 0.0f;
#pragma unroll
  for (int i = 0; i < 4; ++i) {
    int c = t + i * 128;
    float u = xr[c] + (rr ? rr[c] : 0.0f);
    v[i] = u; s += u;
  }
  red[t] = s; __syncthreads();
  for (int o = 64; o > 0; o >>= 1) { if (t < o) red[t] += red[t + o]; __syncthreads(); }
  float mean = red[0] / 512.0f;
  __syncthreads();
  s = 0.0f;
#pragma unroll
  for (int i = 0; i < 4; ++i) { float d = v[i] - mean; s += d * d; }
  red[t] = s; __syncthreads();
  for (int o = 64; o > 0; o >>= 1) { if (t < o) red[t] += red[t + o]; __syncthreads(); }
  float inv = rsqrtf(red[0] / 512.0f + 1e-5f);
#pragma unroll
  for (int i = 0; i < 4; ++i) {
    int c = t + i * 128;
    Y[(long)row * DM + c] = (v[i] - mean) * inv * G[c] + Bt[c];
  }
}

// BatchNorm stats over all rows per channel.
__global__ __launch_bounds__(256)
void bn_stats(const float* __restrict__ Y, float* __restrict__ mu,
              float* __restrict__ var, int Mrows) {
  __shared__ float r1[256], r2[256];
  int c = blockIdx.x, t = threadIdx.x;
  float s = 0.0f, s2 = 0.0f;
  for (int r = t; r < Mrows; r += 256) {
    float v = Y[(long)r * DM + c];
    s += v; s2 += v * v;
  }
  r1[t] = s; r2[t] = s2; __syncthreads();
  for (int o = 128; o > 0; o >>= 1) {
    if (t < o) { r1[t] += r1[t + o]; r2[t] += r2[t + o]; }
    __syncthreads();
  }
  if (t == 0) {
    float m = r1[0] / Mrows;
    mu[c] = m;
    var[c] = r2[0] / Mrows - m * m;
  }
}

__global__ void bn_elu(const float* __restrict__ Y, const float* __restrict__ mu,
                       const float* __restrict__ var, const float* __restrict__ G,
                       const float* __restrict__ Bt, float* __restrict__ Z, long total) {
  long idx = (long)blockIdx.x * 256 + threadIdx.x;
  if (idx >= total) return;
  int c = idx % DM;
  float v = (Y[idx] - mu[c]) * rsqrtf(var[c] + 1e-5f) * G[c] + Bt[c];
  Z[idx] = v > 0.0f ? v : (__expf(v) - 1.0f);
}

// maxpool window 3 stride 2 pad (1,1) over time: [B,L,512] -> [B,L/2,512]
__global__ void maxpool_k(const float* __restrict__ Z, float* __restrict__ O, int L) {
  int Lo = L / 2;
  long idx = (long)blockIdx.x * 256 + threadIdx.x;
  if (idx >= (long)BATCH * Lo * DM) return;
  int b = idx / ((long)Lo * DM);
  int rem = idx % ((long)Lo * DM);
  int tt = rem / DM, c = rem % DM;
  float m = -1e30f;
  for (int j = 0; j < 3; ++j) {
    int st = 2 * tt - 1 + j;
    if (st >= 0 && st < L) m = fmaxf(m, Z[((long)b * L + st) * DM + c]);
  }
  O[idx] = m;
}

// circular pad for conv distill: [B,L,512] -> [B,L+2,512]
__global__ void pad_circ(const float* __restrict__ X, float* __restrict__ Xp, int L) {
  long idx = (long)blockIdx.x * 256 + threadIdx.x;
  if (idx >= (long)BATCH * (L + 2) * DM) return;
  int b = idx / ((long)(L + 2) * DM);
  int rem = idx % ((long)(L + 2) * DM);
  int tt = rem / DM, c = rem % DM;
  int st = tt - 1;
  st = st < 0 ? L - 1 : (st >= L ? 0 : st);
  Xp[idx] = X[((long)b * L + st) * DM + c];
}

// ctx [B,H,L,64] -> [B,L,512]; mix==1 is a flat copy (Informer "mix" quirk).
__global__ void reorder_ctx(const float* __restrict__ Ctx, float* __restrict__ Y,
                            int L, int mix) {
  long idx = (long)blockIdx.x * 256 + threadIdx.x;
  if (idx >= (long)BATCH * L * DM) return;
  if (mix) { Y[idx] = Ctx[idx]; return; }
  int b = idx / ((long)L * DM);
  int rem = idx % ((long)L * DM);
  int l = rem / DM, c = rem % DM;
  int h = c >> 6, d = c & 63;
  Y[idx] = Ctx[(((long)b * NH + h) * L + l) * HD + d];
}

// Final projection on last 512 decoder positions: [., 512] @ [512,7] + b
__global__ void proj_kernel(const float* __restrict__ Dec, const float* __restrict__ Pw,
                            const float* __restrict__ Pb, float* __restrict__ Out) {
  int idx = blockIdx.x * 256 + threadIdx.x;
  if (idx >= BATCH * 512 * 7) return;
  int b = idx / (512 * 7);
  int rem = idx % (512 * 7);
  int tt = rem / 7, c = rem % 7;
  const float* dr = Dec + ((long)b * 1024 + 512 + tt) * DM;
  float s = Pb[c];
  for (int k = 0; k < DM; ++k) s += dr[k] * Pw[k * 7 + c];
  Out[idx] = s;
}

// ---------------------------------------------------------------------------
// Orchestration
// ---------------------------------------------------------------------------
extern "C" void kernel_launch(void* const* d_in, const int* in_sizes, int n_in,
                              void* d_out, int out_size, void* d_ws, size_t ws_size,
                              hipStream_t stream) {
  if (n_in < 120) return;  // expects pytree-flattened params (alphabetical)

  const float* x_enc    = (const float*)d_in[0];
  const int*   mark_enc = (const int*)d_in[1];
  const float* x_dec    = (const float*)d_in[2];
  const int*   mark_dec = (const int*)d_in[3];
  auto P = [&](int i) { return (const float*)d_in[i]; };

  // param leaf indices (jax tree flatten, dict keys sorted):
  // cv[0]=4..7 (b,beta,g,w)  cv[1]=8..11
  // dec[0]=12..37  dec[1]=38..63   (b1,b2,cross{bk,bo,bq,bv,wk,wo,wq,wv},
  //   n1b,n1g,n2b,n2g,n3b,n3g, self{bk..wv}, w1,w2)
  // dec_emb=64  dec_nb=65 dec_ng=66
  // enc[i]=67/83/99 (b1,b2,bk,bo,bq,bv,n1b,n1g,n2b,n2g,w1,w2,wk,wo,wq,wv)
  // enc_emb=115 enc_nb=116 enc_ng=117 proj_b=118 proj_w=119
  const int CV[2]  = {4, 8};
  const int DEC[2] = {12, 38};
  const int ENC[3] = {67, 83, 99};

  float* ws    = (float*)d_ws;
  float* bX0   = ws;                  // 2M floats each
  float* bX1   = bX0 + 2097152;
  float* bNew  = bX1 + 2097152;
  float* bQ    = bNew + 2097152;
  float* bK    = bQ + 2097152;
  float* bV    = bK + 2097152;
  float* bCtx  = bV + 2097152;
  float* bY    = bCtx + 2097152;
  float* bFFN  = bY + 2097152;        // 8M floats
  float* bPad  = bFFN + 8388608;      // 4*1026*512
  float* bStat = bPad + 2101248;      // 1024
  float* bEnc  = bStat + 1024;        // encoder output (<= 2M)

  auto gemm = [&](const float* A, const float* W, const float* bias, float* C,
                  int M, int N, int K, int act, int accum,
                  int aRPB, long aBS, int wRS, int wCS, long wOff) {
    dim3 g(N / 64, M / 128);
    gemm_wmma<<<g, 256, 0, stream>>>(A, W, bias, C, M, N, K, act, accum,
                                     aRPB, aBS, wRS, wCS, wOff);
  };
  auto linear = [&](const float* A, int wq, int bq, float* C, int M, int N, int K,
                    int act) {
    gemm(A, P(wq), P(bq), C, M, N, K, act, 0, M, 0, N, 1, 0);
  };
  auto ew = [&](long total) { return (int)((total + 255) / 256); };

  // ------------------------------- Encoder -------------------------------
  int L = 1024;
  float *x = bX0, *xa = bX1;
  embed_kernel<<<ew((long)BATCH * L * DM), 256, 0, stream>>>(
      x_enc, mark_enc, P(115), x, L, 7);

  for (int li = 0; li < 3; ++li) {
    int base = ENC[li];
    int M = BATCH * L;
    // self-attention (ProbSparse realized as full attention)
    linear(x, base + 14, base + 4, bQ, M, DM, DM, 0);  // wq,bq
    linear(x, base + 12, base + 2, bK, M, DM, DM, 0);  // wk,bk
    linear(x, base + 15, base + 5, bV, M, DM, DM, 0);  // wv,bv
    attn_wmma<<<dim3(L / 64, NH, BATCH), 128, 0, stream>>>(bQ, bK, bV, bCtx, L, L, 0);
    reorder_ctx<<<ew((long)M * DM), 256, 0, stream>>>(bCtx, bY, L, 0);
    linear(bY, base + 13, base + 3, bNew, M, DM, DM, 0);  // wo,bo
    ln_kernel<<<M, 128, 0, stream>>>(bNew, x, P(base + 7), P(base + 6), xa);
    { float* tp = x; x = xa; xa = tp; }
    // FFN
    linear(x, base + 10, base + 0, bFFN, M, DFF, DM, 1);    // w1,b1, gelu
    linear(bFFN, base + 11, base + 1, bNew, M, DM, DFF, 0); // w2,b2
    ln_kernel<<<M, 128, 0, stream>>>(bNew, x, P(base + 9), P(base + 8), xa);
    { float* tp = x; x = xa; xa = tp; }

    if (li < 2) {
      // conv distill: circ conv(512->512,k=3) via 3 accumulated GEMMs
      int cb = CV[li];
      pad_circ<<<ew((long)BATCH * (L + 2) * DM), 256, 0, stream>>>(x, bPad, L);
      for (int dt = 0; dt < 3; ++dt)
        gemm(bPad + (long)dt * DM, P(cb + 3), nullptr, bNew, M, DM, DM, 0,
             dt > 0 ? 1 : 0, L, (long)(L + 2) * DM, 3, 1536, dt);
      bn_stats<<<DM, 256, 0, stream>>>(bNew, bStat, bStat + 512, M);
      bn_elu<<<ew((long)M * DM), 256, 0, stream>>>(bNew, bStat, bStat + 512,
                                                   P(cb + 2), P(cb + 1), bY,
                                                   (long)M * DM);
      maxpool_k<<<ew((long)BATCH * (L / 2) * DM), 256, 0, stream>>>(bY, xa, L);
      { float* tp = x; x = xa; xa = tp; }
      L /= 2;
    }
  }
  // encoder final LN -> bEnc (L == 256)
  ln_kernel<<<BATCH * L, 128, 0, stream>>>(x, nullptr, P(117), P(116), bEnc);
  const int Lenc = L;           // 256
  const int Menc = BATCH * Lenc;

  // ------------------------------- Decoder -------------------------------
  const int Ld = 1024, Md = BATCH * Ld;
  x = bX0; xa = bX1;
  embed_kernel<<<ew((long)Md * DM), 256, 0, stream>>>(x_dec, mark_dec, P(64), x, Ld, 7);

  for (int lj = 0; lj < 2; ++lj) {
    int base = DEC[lj];
    // masked self-attention (mix=True)
    linear(x, base + 22, base + 18, bQ, Md, DM, DM, 0);  // self.wq
    linear(x, base + 20, base + 16, bK, Md, DM, DM, 0);  // self.wk
    linear(x, base + 23, base + 19, bV, Md, DM, DM, 0);  // self.wv
    attn_wmma<<<dim3(Ld / 64, NH, BATCH), 128, 0, stream>>>(bQ, bK, bV, bCtx, Ld, Ld, 1);
    reorder_ctx<<<ew((long)Md * DM), 256, 0, stream>>>(bCtx, bY, Ld, 1);
    linear(bY, base + 21, base + 17, bNew, Md, DM, DM, 0);  // self.wo
    ln_kernel<<<Md, 128, 0, stream>>>(bNew, x, P(base + 11), P(base + 10), xa);
    { float* tp = x; x = xa; xa = tp; }
    // cross attention (full, mix=False)
    linear(x, base + 8, base + 4, bQ, Md, DM, DM, 0);       // cross.wq
    linear(bEnc, base + 6, base + 2, bK, Menc, DM, DM, 0);  // cross.wk
    linear(bEnc, base + 9, base + 5, bV, Menc, DM, DM, 0);  // cross.wv
    attn_wmma<<<dim3(Ld / 64, NH, BATCH), 128, 0, stream>>>(bQ, bK, bV, bCtx, Ld, Lenc, 0);
    reorder_ctx<<<ew((long)Md * DM), 256, 0, stream>>>(bCtx, bY, Ld, 0);
    linear(bY, base + 7, base + 3, bNew, Md, DM, DM, 0);    // cross.wo
    ln_kernel<<<Md, 128, 0, stream>>>(bNew, x, P(base + 13), P(base + 12), xa);
    { float* tp = x; x = xa; xa = tp; }
    // FFN
    linear(x, base + 24, base + 0, bFFN, Md, DFF, DM, 1);
    linear(bFFN, base + 25, base + 1, bNew, Md, DM, DFF, 0);
    ln_kernel<<<Md, 128, 0, stream>>>(bNew, x, P(base + 15), P(base + 14), xa);
    { float* tp = x; x = xa; xa = tp; }
  }
  // final LN + projection of last 512 positions
  ln_kernel<<<Md, 128, 0, stream>>>(x, nullptr, P(66), P(65), bY);
  proj_kernel<<<(BATCH * 512 * 7 + 255) / 256, 256, 0, stream>>>(
      bY, P(119), P(118), (float*)d_out);
}
